// MambaLayer_3126736191494
// MI455X (gfx1250) — compile-verified, matrix operand order checked by reading, not run on hardware
//
#include <hip/hip_runtime.h>
#include <hip/hip_bf16.h>
#include <math.h>

// ---------------- types for WMMA ----------------
typedef __bf16 bf16_t;
typedef bf16_t v16bf __attribute__((ext_vector_type(16)));
typedef float  v8f   __attribute__((ext_vector_type(8)));
typedef unsigned short u16x8 __attribute__((ext_vector_type(8)));

// ---------------- problem constants ----------------
#define DIMC     512
#define D_INNER  1024
#define HEADDIM  128
#define NHEADS   8
#define D_STATE  128
#define D_CONV   4
#define CONV_DIM 1280            // D_INNER + 2*D_STATE
#define D_INPROJ 2312            // 2*D_INNER + 2*D_STATE + NHEADS
#define D_INPROJ_PAD 2368        // padded to multiple of 64
#define BSZ      4
#define SEQL     4096            // 16*16*16
#define MROWS    (BSZ*SEQL)      // 16384

// GEMM tiling: block 128(M) x 64(N), 8 waves as 4(M) x 2(N), wave tile 32x32
#define BM  128
#define BN  64
#define KT  32
#define LDK 40                   // padded LDS row stride (bf16 elems), 80B rows

// ---------------- CDNA5 async-copy helpers ----------------
__device__ __forceinline__ unsigned lds_off(const void* p) {
    return (unsigned)(size_t)p;          // low 32 bits of flat shared ptr = LDS offset
}
__device__ __forceinline__ void async_b128(unsigned lds, const bf16_t* g) {
    asm volatile("global_load_async_to_lds_b128 %0, %1, off"
                 :: "v"(lds), "v"(g) : "memory");
}
__device__ __forceinline__ void wait_async3() {   // <=3 outstanding (next tile in flight)
    asm volatile("s_wait_asynccnt 0x3" ::: "memory");
}
__device__ __forceinline__ void wait_async0() {
    asm volatile("s_wait_asynccnt 0x0" ::: "memory");
}

// ---------------- helpers ----------------
__device__ __forceinline__ v16bf frag_b16(const bf16_t* rowptr, int kb) {
    union { v16bf v; u16x8 h[2]; } f;
    f.h[0] = *reinterpret_cast<const u16x8*>(rowptr + kb);
    f.h[1] = *reinterpret_cast<const u16x8*>(rowptr + kb + 16);
    return f.v;
}
__device__ __forceinline__ float silu_f(float v) {
    return v * (1.0f / (1.0f + __expf(-v)));
}

// ---------------- 1) transpose x(B,512,L) -> u_bf(B*L,512) in bf16 ----------------
__global__ __launch_bounds__(256) void transpose_kernel(const float* __restrict__ x,
                                                        bf16_t* __restrict__ u) {
    __shared__ float t[32][33];
    const int b  = blockIdx.z;
    const int l0 = blockIdx.x * 32;
    const int d0 = blockIdx.y * 32;
#pragma unroll
    for (int j = 0; j < 4; ++j) {
        int d = d0 + threadIdx.y + j * 8;
        int l = l0 + threadIdx.x;
        t[threadIdx.y + j * 8][threadIdx.x] = x[((size_t)b * DIMC + d) * SEQL + l];
    }
    __syncthreads();
#pragma unroll
    for (int j = 0; j < 4; ++j) {
        int l = l0 + threadIdx.y + j * 8;
        int d = d0 + threadIdx.x;
        u[((size_t)b * SEQL + l) * DIMC + d] = (bf16_t)t[threadIdx.x][threadIdx.y + j * 8];
    }
}

// ---------------- 1b) fp32 -> bf16 weight convert with row zero-padding ----------------
__global__ __launch_bounds__(256) void convert_pad_kernel(const float* __restrict__ src,
                                                          bf16_t* __restrict__ dst,
                                                          int rows, int cols /*mult of 4*/) {
    size_t e = ((size_t)blockIdx.x * 256 + threadIdx.x) * 4;
    int row = (int)(e / cols);
    if (row < rows) {
        float4 f = *reinterpret_cast<const float4*>(src + e);
        dst[e + 0] = (bf16_t)f.x; dst[e + 1] = (bf16_t)f.y;
        dst[e + 2] = (bf16_t)f.z; dst[e + 3] = (bf16_t)f.w;
    } else {
        dst[e + 0] = (bf16_t)0.f; dst[e + 1] = (bf16_t)0.f;
        dst[e + 2] = (bf16_t)0.f; dst[e + 3] = (bf16_t)0.f;
    }
}

// ---------------- 2)/6) bf16 WMMA GEMM, async double-buffered tiles ----------------
// Out[m,n] = sum_k A[m,k]*W[n,k]; A: MxK bf16, W: NPADxK bf16 (rows >= gridDim.x*BN)
// MODE 0: Out row-major (M x N, fp32). MODE 1: transpose-store + residual to outT[b,n,l].
template <int MODE>
__global__ __launch_bounds__(256) void gemm_wmma_bf16(const bf16_t* __restrict__ A,
                                                      const bf16_t* __restrict__ W,
                                                      float* __restrict__ Out,
                                                      int M, int N, int K,
                                                      const float* __restrict__ resid,
                                                      float* __restrict__ outT) {
    __shared__ __align__(16) bf16_t As[2][BM * LDK];
    __shared__ __align__(16) bf16_t Ws[2][BN * LDK];

    const int tid  = threadIdx.x;
    const int m0   = blockIdx.y * BM;
    const int n0   = blockIdx.x * BN;
    const int wave = tid >> 5;
    const int lane = tid & 31;
    const int wm   = wave & 3;             // 4 waves along M (32 rows each)
    const int wn   = wave >> 2;            // 2 waves along N (32 cols each)
    const int r    = lane & 15;
    const int kb   = (lane >> 4) << 3;     // 0 or 8

    // tile-fill assignment: 16B segments, K-contiguous
    const int frow = tid >> 2;             // 0..63
    const int fseg = (tid & 3) * 8;        // bf16 element offset: 0,8,16,24

    // incrementally-stepped async source pointers (advance by KT per stage)
    const bf16_t* pa0 = A + (size_t)(m0 + frow) * K + fseg;        // rows 0..63
    const bf16_t* pa1 = A + (size_t)(m0 + 64 + frow) * K + fseg;   // rows 64..127
    const bf16_t* pw  = W + (size_t)(n0 + frow) * K + fseg;

    unsigned lA0[2], lA1[2], lW[2];
#pragma unroll
    for (int bu = 0; bu < 2; ++bu) {
        lA0[bu] = lds_off(&As[bu][frow * LDK + fseg]);
        lA1[bu] = lds_off(&As[bu][(64 + frow) * LDK + fseg]);
        lW[bu]  = lds_off(&Ws[bu][frow * LDK + fseg]);
    }

    v8f acc00 = {}, acc01 = {}, acc10 = {}, acc11 = {};

    auto compute = [&](int cur) {
        v16bf a0 = frag_b16(&As[cur][(wm * 32 + r) * LDK], kb);
        v16bf a1 = frag_b16(&As[cur][(wm * 32 + 16 + r) * LDK], kb);
        v16bf b0 = frag_b16(&Ws[cur][(wn * 32 + r) * LDK], kb);
        v16bf b1 = frag_b16(&Ws[cur][(wn * 32 + 16 + r) * LDK], kb);
        acc00 = __builtin_amdgcn_wmma_f32_16x16x32_bf16(false, a0, false, b0, (short)0, acc00, false, false);
        acc01 = __builtin_amdgcn_wmma_f32_16x16x32_bf16(false, a0, false, b1, (short)0, acc01, false, false);
        acc10 = __builtin_amdgcn_wmma_f32_16x16x32_bf16(false, a1, false, b0, (short)0, acc10, false, false);
        acc11 = __builtin_amdgcn_wmma_f32_16x16x32_bf16(false, a1, false, b1, (short)0, acc11, false, false);
    };

    const int nk = K / KT;
    // prologue: stage tile 0
    async_b128(lA0[0], pa0);
    async_b128(lA1[0], pa1);
    async_b128(lW[0],  pw);

    // steady state: unconditionally prefetch tile i+1, compute tile i
    for (int i = 0; i < nk - 1; ++i) {
        const int cur = i & 1;
        const int nxt = cur ^ 1;
        pa0 += KT; pa1 += KT; pw += KT;
        async_b128(lA0[nxt], pa0);
        async_b128(lA1[nxt], pa1);
        async_b128(lW[nxt],  pw);
        wait_async3();                     // tile i landed (in-order), tile i+1 in flight
        __syncthreads();
        compute(cur);
        __syncthreads();                   // readers done before buffer reuse
    }
    // epilogue: last tile
    wait_async0();
    __syncthreads();
    compute((nk - 1) & 1);

    // store: acc elem i -> row = base + i + (lane>=16 ? 8:0), col = lane&15
    const int mbase = m0 + wm * 32 + (lane >> 4) * 8;
    const int nbase = n0 + wn * 32 + (lane & 15);
    v8f accs[2][2] = {{acc00, acc01}, {acc10, acc11}};
#pragma unroll
    for (int a = 0; a < 2; ++a) {
#pragma unroll
        for (int j = 0; j < 2; ++j) {
            const int n = nbase + j * 16;
#pragma unroll
            for (int i = 0; i < 8; ++i) {
                const int m = mbase + a * 16 + i;
                if (MODE == 0) {
                    if (n < N) Out[(size_t)m * N + n] = accs[a][j][i];
                } else {
                    int b = m >> 12;               // m / SEQL
                    int l = m & (SEQL - 1);
                    size_t o = ((size_t)b * DIMC + n) * SEQL + l;
                    outT[o] = accs[a][j][i] + resid[o];
                }
            }
        }
    }
}

// ---------------- 3) causal depthwise conv + SiLU, softplus(dt) ----------------
__global__ __launch_bounds__(256) void conv_dt_kernel(const float* __restrict__ zx,
                                                      const float* __restrict__ conv_w,
                                                      const float* __restrict__ conv_b,
                                                      const float* __restrict__ dt_bias,
                                                      float* __restrict__ xbc,
                                                      float* __restrict__ dtb) {
    const int l = blockIdx.x;
    const int b = blockIdx.y;
    const size_t row = (size_t)b * SEQL + l;

    for (int c = threadIdx.x; c < CONV_DIM; c += 256) {
        float v = conv_b[c];
#pragma unroll
        for (int k = 0; k < D_CONV; ++k) {
            int ll = l + k - (D_CONV - 1);
            if (ll >= 0)
                v += zx[((size_t)b * SEQL + ll) * D_INPROJ + D_INNER + c] * conv_w[c * D_CONV + k];
        }
        xbc[row * CONV_DIM + c] = silu_f(v);
    }
    if (threadIdx.x < NHEADS) {
        int h = threadIdx.x;
        float raw = zx[row * D_INPROJ + (D_INNER + CONV_DIM) + h] + dt_bias[h];
        float sp  = (raw > 20.0f) ? raw : log1pf(__expf(raw));
        dtb[row * NHEADS + h] = sp;
    }
}

// ---------------- 4) SSM scan: (b,h,p-chunk) blocks, state in registers ----------------
// grid (NHEADS, BSZ, 4): each block owns 32 of 128 head-dim rows.
// 256 threads = 32 p-rows x 8 n-chunks of 16 states each.
__global__ __launch_bounds__(256) void scan_kernel(const float* __restrict__ xbc,
                                                   const float* __restrict__ dtb,
                                                   const float* __restrict__ A_log,
                                                   const float* __restrict__ D_param,
                                                   float* __restrict__ ybuf) {
    const int h  = blockIdx.x;
    const int b  = blockIdx.y;
    const int pc = blockIdx.z;
    const int tid = threadIdx.x;
    const int pl  = tid >> 3;            // 0..31
    const int nch = tid & 7;             // 0..7
    const int nb  = nch * 16;
    const int p   = pc * 32 + pl;

    const float Ah = -__expf(A_log[h]);
    const float Dp = D_param[h];

    float st[16];
#pragma unroll
    for (int i = 0; i < 16; ++i) st[i] = 0.0f;

    __shared__ float Bs[D_STATE];
    __shared__ float Cs[D_STATE];

    const size_t base = (size_t)b * SEQL;
    for (int l = 0; l < SEQL; ++l) {
        const size_t row = base + l;
        if (tid < 128) Bs[tid]       = xbc[row * CONV_DIM + D_INNER + tid];
        else           Cs[tid - 128] = xbc[row * CONV_DIM + D_INNER + D_STATE + (tid - 128)];
        const float dt = dtb[row * NHEADS + h];
        const float xv = xbc[row * CONV_DIM + h * HEADDIM + p];
        __syncthreads();

        const float dA = __expf(dt * Ah);
        const float cf = dt * xv;
        float y = 0.0f;
#pragma unroll
        for (int n = 0; n < 16; ++n) {
            float s = st[n] * dA + cf * Bs[nb + n];
            st[n] = s;
            y += s * Cs[nb + n];
        }
        y += __shfl_xor(y, 1, 32);
        y += __shfl_xor(y, 2, 32);
        y += __shfl_xor(y, 4, 32);
        if (nch == 0)
            ybuf[row * D_INNER + h * HEADDIM + p] = y + Dp * xv;
        __syncthreads();
    }
}

// ---------------- 5) gate with SiLU(z) + RMSNorm -> bf16 ----------------
__global__ __launch_bounds__(256) void gatenorm_kernel(const float* __restrict__ zx,
                                                       const float* __restrict__ norm_w,
                                                       const float* __restrict__ ybuf,
                                                       bf16_t* __restrict__ ynorm) {
    const size_t row = (size_t)blockIdx.y * SEQL + blockIdx.x;
    const int tid = threadIdx.x;

    float vals[4];
    float ss = 0.0f;
#pragma unroll
    for (int j = 0; j < 4; ++j) {
        int e = tid + j * 256;
        float yv = ybuf[row * D_INNER + e];
        float zv = zx[row * D_INPROJ + e];
        yv *= silu_f(zv);
        vals[j] = yv;
        ss += yv * yv;
    }
#pragma unroll
    for (int off = 16; off >= 1; off >>= 1) ss += __shfl_xor(ss, off, 32);

    __shared__ float red[8];
    __shared__ float s_inv;
    if ((tid & 31) == 0) red[tid >> 5] = ss;
    __syncthreads();
    if (tid == 0) {
        float tot = 0.0f;
#pragma unroll
        for (int w = 0; w < 8; ++w) tot += red[w];
        s_inv = rsqrtf(tot / (float)D_INNER + 1e-5f);
    }
    __syncthreads();
    const float inv = s_inv;
#pragma unroll
    for (int j = 0; j < 4; ++j) {
        int e = tid + j * 256;
        ynorm[row * D_INNER + e] = (bf16_t)(vals[j] * inv * norm_w[e]);
    }
}

// ---------------- launch ----------------
extern "C" void kernel_launch(void* const* d_in, const int* in_sizes, int n_in,
                              void* d_out, int out_size, void* d_ws, size_t ws_size,
                              hipStream_t stream) {
    const float* x          = (const float*)d_in[0];
    const float* in_proj_w  = (const float*)d_in[1];
    const float* conv_w     = (const float*)d_in[2];
    const float* conv_b     = (const float*)d_in[3];
    const float* dt_bias    = (const float*)d_in[4];
    const float* A_log      = (const float*)d_in[5];
    const float* D_param    = (const float*)d_in[6];
    const float* norm_w     = (const float*)d_in[7];
    const float* out_proj_w = (const float*)d_in[8];

    char* ws = (char*)d_ws;
    size_t off = 0;
    bf16_t* u_bf   = (bf16_t*)(ws + off); off += (size_t)MROWS * DIMC * 2;         //  16.8 MB
    float*  zx     = (float*) (ws + off); off += (size_t)MROWS * D_INPROJ * 4;     // 151.5 MB
    float*  xbc    = (float*) (ws + off); off += (size_t)MROWS * CONV_DIM * 4;     //  83.9 MB
    float*  dtb    = (float*) (ws + off); off += (size_t)MROWS * NHEADS * 4;       //   0.5 MB
    float*  ybuf   = (float*) (ws + off); off += (size_t)MROWS * D_INNER * 4;      //  67.1 MB
    bf16_t* yn_bf  = (bf16_t*)(ws + off); off += (size_t)MROWS * D_INNER * 2;      //  33.6 MB
    bf16_t* wi_bf  = (bf16_t*)(ws + off); off += (size_t)D_INPROJ_PAD * DIMC * 2;  //   2.4 MB
    bf16_t* wo_bf  = (bf16_t*)(ws + off); off += (size_t)DIMC * D_INNER * 2;       //   1.0 MB

    // 1) u_bf = bf16(transpose(x)); weights -> bf16 (in_proj padded to 2368 rows)
    transpose_kernel<<<dim3(SEQL / 32, DIMC / 32, BSZ), dim3(32, 8), 0, stream>>>(x, u_bf);
    convert_pad_kernel<<<(D_INPROJ_PAD * DIMC) / 1024, 256, 0, stream>>>(
        in_proj_w, wi_bf, D_INPROJ, DIMC);
    convert_pad_kernel<<<(DIMC * D_INNER) / 1024, 256, 0, stream>>>(
        out_proj_w, wo_bf, DIMC, D_INNER);

    // 2) zxbcdt = u @ in_proj_w^T  (M=16384, N=2312, K=512)
    gemm_wmma_bf16<0><<<dim3(D_INPROJ_PAD / BN, MROWS / BM), 256, 0, stream>>>(
        u_bf, wi_bf, zx, MROWS, D_INPROJ, DIMC, nullptr, nullptr);

    // 3) conv + SiLU, softplus(dt)
    conv_dt_kernel<<<dim3(SEQL, BSZ), 256, 0, stream>>>(zx, conv_w, conv_b, dt_bias, xbc, dtb);

    // 4) SSM scan (128 independent workgroups)
    scan_kernel<<<dim3(NHEADS, BSZ, 4), 256, 0, stream>>>(xbc, dtb, A_log, D_param, ybuf);

    // 5) gate + RMSNorm -> bf16
    gatenorm_kernel<<<dim3(SEQL, BSZ), 256, 0, stream>>>(zx, norm_w, ybuf, yn_bf);

    // 6) out = y @ out_proj_w^T, transpose-store + residual  (M=16384, N=512, K=1024)
    gemm_wmma_bf16<1><<<dim3(DIMC / BN, MROWS / BM), 256, 0, stream>>>(
        yn_bf, wo_bf, nullptr, MROWS, DIMC, D_INNER, x, (float*)d_out);
}
